// LlamaAttention_33389075759402
// MI455X (gfx1250) — matrix operand order repaired and checked by reading, not compiled, not run on hardware
//
#include <hip/hip_runtime.h>
#include <hip/hip_bf16.h>

// ---------------------------------------------------------------------------
// LlamaAttention decode step for MI455X (gfx1250), fp32 I/O, f16 WMMA compute.
// KV streaming: K chunks via TDM (tensor_load_to_lds, TENSORcnt),
//               V chunks via global_load_async_to_lds_b128 (ASYNCcnt),
//               double-buffered in LDS.
// B=4, S=16, H=4096, NH=32, NKV=8, HD=128, PAST=8192, KTOT=8208.
// ---------------------------------------------------------------------------

#define B_    4
#define S_    16
#define H_    4096
#define NH_   32
#define NKV_  8
#define HD_   128
#define PAST_ 8192
#define KTOT_ (PAST_ + S_)       // 8208
#define ROWS_ (B_ * S_)          // 64

typedef __attribute__((ext_vector_type(16))) _Float16 v16h;
typedef __attribute__((ext_vector_type(8)))  float    v8f;
typedef __attribute__((ext_vector_type(4)))  unsigned int v4u;
typedef __attribute__((ext_vector_type(8)))  int      v8i;
typedef __attribute__((ext_vector_type(4)))  int      v4i;

union AFrag { v16h v; _Float16 h[16]; };
union BFrag { v16h v; _Float16 h[16]; };
union CFrag { v8f  v; float     f[8]; };

__device__ __forceinline__ float halfmax16(float v) {
    v = fmaxf(v, __shfl_xor(v, 1));
    v = fmaxf(v, __shfl_xor(v, 2));
    v = fmaxf(v, __shfl_xor(v, 4));
    v = fmaxf(v, __shfl_xor(v, 8));
    return v;
}
__device__ __forceinline__ float halfsum16(float v) {
    v += __shfl_xor(v, 1);
    v += __shfl_xor(v, 2);
    v += __shfl_xor(v, 4);
    v += __shfl_xor(v, 8);
    return v;
}

__device__ __forceinline__ uint32_t lds_off_of(const void* p) {
    return (uint32_t)(uintptr_t)p;   // flat LDS addr truncates to LDS offset
}

// Async DMA: global -> LDS, 16 bytes per lane, tracked by ASYNCcnt.
__device__ __forceinline__ void async_copy_b128(uint32_t lds_off, const void* gptr) {
    asm volatile("global_load_async_to_lds_b128 %0, %1, off"
                 :: "v"(lds_off), "v"((uint64_t)(uintptr_t)gptr)
                 : "memory");
}
__device__ __forceinline__ void wait_async0() {
    asm volatile("s_wait_asynccnt 0x0" ::: "memory");
}

// ---------------------------------------------------------------------------
// TDM: one-instruction 2D tile DMA (rows x 128 fp32, row stride in elements)
// from global memory into LDS. Tracked by TENSORcnt.  D# packing per CDNA5
// ISA ch.8 (group0 128b, group1 256b; 2D -> groups 2/3 zero).
// ---------------------------------------------------------------------------
#if __has_builtin(__builtin_amdgcn_tensor_load_to_lds)
#define HAVE_TDM_ 1
__device__ __forceinline__ void tdm_load_2d_f32(uint32_t lds_off, const void* gptr,
                                                uint32_t rows, uint32_t row_stride)
{
    const uint64_t ga = (uint64_t)(uintptr_t)gptr;
    v4u g0;
    g0.x = 1u;                                   // count=1, is_restore=0, gather=0
    g0.y = lds_off;                              // lds_addr (bytes)
    g0.z = (uint32_t)ga;                         // global_addr[31:0]
    g0.w = (uint32_t)(ga >> 32) | (2u << 30);    // global_addr[56:32] | type=2
    v8i g1;
    g1[0] = (int)(2u << 16);                     // data_size=2 (4B), wg_mask=0
    g1[1] = (int)(128u << 16);                   // tensor_dim0[15:0]=128
    g1[2] = (int)(rows << 16);                   // tensor_dim1[15:0]=rows
    g1[3] = (int)(128u << 16);                   // tile_dim0=128
    g1[4] = (int)rows;                           // tile_dim1=rows, tile_dim2=0
    g1[5] = (int)row_stride;                     // tensor_dim0_stride[31:0]
    g1[6] = 0;
    g1[7] = 0;
    const v4i z4 = {0, 0, 0, 0};
#if defined(__clang_major__) && (__clang_major__ >= 23)
    const v8i z8 = {0, 0, 0, 0, 0, 0, 0, 0};
    __builtin_amdgcn_tensor_load_to_lds(g0, g1, z4, z4, z8, 0);
#else
    __builtin_amdgcn_tensor_load_to_lds(g0, g1, z4, z4, 0);
#endif
}
#else
#define HAVE_TDM_ 0
#endif

// ---------------------------------------------------------------------------
// Generic 64xKdim (row-major, lda=Kdim) @ Kdim x Ncols strip GEMM.
// One wave computes a 64x16 output strip at columns [col0, col0+16).
// ---------------------------------------------------------------------------
__device__ __forceinline__ void gemm_strip_64(
    const float* __restrict__ A, int kdim,
    const float* __restrict__ W, int ldw, int col0,
    float* __restrict__ out, int ldo)
{
    const int lane = threadIdx.x & 31;
    const int n = lane & 15;          // column within tile / A row (M)
    const int g = lane >> 4;          // lane group

    CFrag c[4];
#pragma unroll
    for (int mt = 0; mt < 4; ++mt)
#pragma unroll
        for (int r = 0; r < 8; ++r) c[mt].f[r] = 0.0f;

    for (int k0 = 0; k0 < kdim; k0 += 32) {
        // Pull the weight stream ahead of the strided B gathers.
        if (k0 + 64 < kdim)
            __builtin_prefetch(&W[(size_t)(k0 + 64 + g * 16) * ldw + col0 + n], 0, 1);

        // B fragment: B[k][n] = W[k0+k][col0+n]; lane holds K = g*8 + j (j<8), 16 + g*8 + (j-8) (j>=8)
        BFrag b;
#pragma unroll
        for (int j = 0; j < 16; ++j)
            b.h[j] = (_Float16)W[(size_t)(k0 + (j < 8 ? g * 8 + j : 16 + g * 8 + (j - 8))) * ldw + col0 + n];

#pragma unroll
        for (int mt = 0; mt < 4; ++mt) {
            // A fragment: row m = lane&15 of this 16-row tile
            const float* ar = A + (size_t)(mt * 16 + n) * kdim + k0;
            AFrag a;
#pragma unroll
            for (int j = 0; j < 8; ++j) a.h[j]     = (_Float16)ar[g * 8 + j];
#pragma unroll
            for (int j = 0; j < 8; ++j) a.h[8 + j] = (_Float16)ar[16 + g * 8 + j];
            c[mt].v = __builtin_amdgcn_wmma_f32_16x16x32_f16(
                false, a.v, false, b.v, (short)0, c[mt].v, false, false);
        }
    }

#pragma unroll
    for (int mt = 0; mt < 4; ++mt)
#pragma unroll
        for (int r = 0; r < 8; ++r)
            out[(size_t)(mt * 16 + r + 8 * g) * ldo + col0 + n] = c[mt].f[r];
}

// ---------------------------------------------------------------------------
// Kernel 1: fused QKV projection. 384 strips: 256 (Q) + 64 (K) + 64 (V).
// ---------------------------------------------------------------------------
__global__ __launch_bounds__(128) void qkv_proj_kernel(
    const float* __restrict__ X,
    const float* __restrict__ Wq, const float* __restrict__ Wk,
    const float* __restrict__ Wv,
    float* __restrict__ Qb, float* __restrict__ Kb, float* __restrict__ Vb)
{
    const int wid = blockIdx.x * 4 + (threadIdx.x >> 5);   // 0..383
    if (wid < 256) {
        gemm_strip_64(X, H_, Wq, NH_ * HD_, wid * 16, Qb, NH_ * HD_);
    } else if (wid < 320) {
        gemm_strip_64(X, H_, Wk, NKV_ * HD_, (wid - 256) * 16, Kb, NKV_ * HD_);
    } else {
        gemm_strip_64(X, H_, Wv, NKV_ * HD_, (wid - 320) * 16, Vb, NKV_ * HD_);
    }
}

// ---------------------------------------------------------------------------
// Kernel 2: RoPE (half-split rotate) applied in place to Qb and Kb.
// ---------------------------------------------------------------------------
__global__ void rope_kernel(float* __restrict__ Qb, float* __restrict__ Kb)
{
    const int i = blockIdx.x * blockDim.x + threadIdx.x;   // 0..163839
    float* p;
    int dp, s;
    if (i < ROWS_ * NH_ * (HD_ / 2)) {                     // 131072 Q pairs
        const int r   = i >> 11;
        const int rem = i & 2047;
        const int h   = rem >> 6;
        dp = rem & 63;
        s  = r & 15;
        p  = Qb + (size_t)r * (NH_ * HD_) + h * HD_ + dp;
    } else {
        const int j   = i - ROWS_ * NH_ * (HD_ / 2);
        const int r   = j >> 9;
        const int rem = j & 511;
        const int h   = rem >> 6;
        dp = rem & 63;
        s  = r & 15;
        p  = Kb + (size_t)r * (NKV_ * HD_) + h * HD_ + dp;
    }
    const float pos = (float)(PAST_ + s);
    const float inv = __expf(-0.14391156652693732f * (float)dp);  // 10000^(-dp/64)
    const float ang = pos * inv;
    const float cs = cosf(ang), sn = sinf(ang);
    const float x0 = p[0], x1 = p[64];
    p[0]  = x0 * cs - x1 * sn;
    p[64] = x1 * cs + x0 * sn;
}

// ---------------------------------------------------------------------------
// Kernel 3: flash-decode attention, TDM + async double-buffered KV streaming.
// Block = (b, kvh): 4 waves = the 4 GQA query heads sharing that KV head.
// ---------------------------------------------------------------------------
__global__ __launch_bounds__(128, 1) void attn_kernel(
    const float* __restrict__ Qb, const float* __restrict__ Kb,
    const float* __restrict__ Vb,
    const float* __restrict__ kc, const float* __restrict__ vc,
    float* __restrict__ attn)
{
    __shared__ float    Ksm[2][32 * HD_];     // 2 x 16 KB
    __shared__ float    Vsm[2][32 * HD_];     // 2 x 16 KB
    __shared__ _Float16 Psm[4][16 * 32];      // 4 KB (per-wave P transpose)

    const int tid  = threadIdx.x;
    const int wave = tid >> 5;
    const int lane = tid & 31;
    const int n = lane & 15;                  // col within tile / row m of A
    const int g = lane >> 4;

    const int b   = blockIdx.x >> 3;
    const int kvh = blockIdx.x & 7;
    const int qh  = kvh * 4 + wave;

    const int NCHUNK = (KTOT_ + 31) / 32;     // 257

    // Stage one 32-position KV chunk into LDS buffer `buf`.
    auto issue_chunk = [&](int ch, int buf) {
        const int kpos0 = ch * 32;
        if (kpos0 + 32 <= PAST_) {
            const size_t base =
                ((size_t)(b * PAST_ + kpos0) * NKV_ + kvh) * HD_;
#if HAVE_TDM_
            // K tile: one TDM descriptor, 32 rows x 128 fp32, stride 1024 elts.
            if (wave == 0)
                tdm_load_2d_f32(lds_off_of(&Ksm[buf][0]), kc + base,
                                32u, (uint32_t)(NKV_ * HD_));
#else
#pragma unroll
            for (int i = 0; i < 8; ++i) {
                const int idx = tid + i * 128;
                const size_t off = base + (size_t)(idx >> 5) * (NKV_ * HD_)
                                 + (idx & 31) * 4;
                async_copy_b128(lds_off_of(&Ksm[buf][idx * 4]), kc + off);
            }
#endif
            // V tile: per-wave async DMA, 512B-contiguous row per wave-issue.
#pragma unroll
            for (int i = 0; i < 8; ++i) {
                const int idx = tid + i * 128;        // float4 index, 0..1023
                const size_t off = base + (size_t)(idx >> 5) * (NKV_ * HD_)
                                 + (idx & 31) * 4;
                async_copy_b128(lds_off_of(&Vsm[buf][idx * 4]), vc + off);
            }
        } else {
            // Tail chunk: mix of cache / new-KV / zero padding, synchronous.
            for (int i = tid; i < 1024; i += 128) {
                const int pos  = i >> 5;
                const int d4   = i & 31;
                const int kpos = kpos0 + pos;
                float4 kv4 = make_float4(0.f, 0.f, 0.f, 0.f), vv4 = kv4;
                if (kpos < PAST_) {
                    const size_t off =
                        ((size_t)(b * PAST_ + kpos) * NKV_ + kvh) * HD_;
                    kv4 = ((const float4*)(kc + off))[d4];
                    vv4 = ((const float4*)(vc + off))[d4];
                } else if (kpos < KTOT_) {
                    const size_t off =
                        (size_t)(b * S_ + (kpos - PAST_)) * (NKV_ * HD_) + kvh * HD_;
                    kv4 = ((const float4*)(Kb + off))[d4];
                    vv4 = ((const float4*)(Vb + off))[d4];
                }
                ((float4*)&Ksm[buf][0])[i] = kv4;
                ((float4*)&Vsm[buf][0])[i] = vv4;
            }
        }
    };

    // Q A-fragments (RoPE'd), 1/sqrt(HD) folded in. Row m = token = lane&15.
    const float scale = 0.08838834764831845f;
    const float* qrow = Qb + (size_t)(b * S_ + n) * (NH_ * HD_) + qh * HD_;
    AFrag qa[4];
#pragma unroll
    for (int ds = 0; ds < 4; ++ds) {
        const int d0 = ds * 32;
#pragma unroll
        for (int j = 0; j < 8; ++j)
            qa[ds].h[j]     = (_Float16)(qrow[d0 + g * 8 + j] * scale);
#pragma unroll
        for (int j = 0; j < 8; ++j)
            qa[ds].h[8 + j] = (_Float16)(qrow[d0 + 16 + g * 8 + j] * scale);
    }

    CFrag o[8];
#pragma unroll
    for (int dt = 0; dt < 8; ++dt)
#pragma unroll
        for (int r = 0; r < 8; ++r) o[dt].f[r] = 0.0f;
    float mrun[8], lrun[8];
#pragma unroll
    for (int r = 0; r < 8; ++r) { mrun[r] = -1e30f; lrun[r] = 0.0f; }

    issue_chunk(0, 0);

    for (int ch = 0; ch < NCHUNK; ++ch) {
        const int cur = ch & 1;

        wait_async0();                 // this wave's V DMA for chunk `ch`
#if HAVE_TDM_
        if (wave == 0)
            __builtin_amdgcn_s_wait_tensorcnt(0);   // wave0's K TDM done
#endif
        __syncthreads();               // publish LDS to all waves; prev compute done

        if (ch + 1 < NCHUNK) issue_chunk(ch + 1, cur ^ 1);   // overlap next load

        const float* Kc = &Ksm[cur][0];
        const float* Vc = &Vsm[cur][0];

        // Scores: two 16x16 tiles (k-positions n and 16+n), d reduced 32/step
        CFrag c0, c1;
#pragma unroll
        for (int r = 0; r < 8; ++r) { c0.f[r] = 0.0f; c1.f[r] = 0.0f; }
#pragma unroll
        for (int ds = 0; ds < 4; ++ds) {
            const int d0 = ds * 32;
            BFrag bk0, bk1;
#pragma unroll
            for (int j = 0; j < 16; ++j) {
                bk0.h[j] = (_Float16)Kc[n * HD_        + d0 + (j < 8 ? g * 8 + j : 16 + g * 8 + (j - 8))];
                bk1.h[j] = (_Float16)Kc[(16 + n) * HD_ + d0 + (j < 8 ? g * 8 + j : 16 + g * 8 + (j - 8))];
            }
            c0.v = __builtin_amdgcn_wmma_f32_16x16x32_f16(
                false, qa[ds].v, false, bk0.v, (short)0, c0.v, false, false);
            c1.v = __builtin_amdgcn_wmma_f32_16x16x32_f16(
                false, qa[ds].v, false, bk1.v, (short)0, c1.v, false, false);
        }

        // Bottom-right causal mask: only the final (new-token) chunk is partial.
        if (ch == NCHUNK - 1) {
#pragma unroll
            for (int r = 0; r < 8; ++r) {
                const int m = r + 8 * g;
                if (n > m) c0.f[r] = -1e30f;  // kpos = PAST+n allowed iff n <= m
                c1.f[r] = -1e30f;             // kpos >= KTOT: never allowed
            }
        }

        // Online softmax update (rows live in 16-lane halves)
        float alpha[8];
#pragma unroll
        for (int r = 0; r < 8; ++r) {
            float cm = halfmax16(fmaxf(c0.f[r], c1.f[r]));
            const float nm = fmaxf(mrun[r], cm);
            const float al = __expf(mrun[r] - nm);
            const float p0 = __expf(c0.f[r] - nm);
            const float p1 = __expf(c1.f[r] - nm);
            lrun[r] = lrun[r] * al + halfsum16(p0 + p1);
            mrun[r] = nm;
            alpha[r] = al;
            const int m = r + 8 * g;
            Psm[wave][m * 32 + n]      = (_Float16)p0;
            Psm[wave][m * 32 + 16 + n] = (_Float16)p1;
        }
#pragma unroll
        for (int dt = 0; dt < 8; ++dt)
#pragma unroll
            for (int r = 0; r < 8; ++r) o[dt].f[r] *= alpha[r];

        // Intra-wave LDS transpose barrier (C-layout -> A-layout)
        asm volatile("s_wait_dscnt 0" ::: "memory");

        AFrag pa;                             // P as 16x32 A fragment, row = lane&15
#pragma unroll
        for (int j = 0; j < 8; ++j) pa.h[j]     = Psm[wave][n * 32 + g * 8 + j];
#pragma unroll
        for (int j = 0; j < 8; ++j) pa.h[8 + j] = Psm[wave][n * 32 + 16 + g * 8 + j];

        // O += P @ V : 8 d-tiles of 16, Kdim = 32 chunk positions
#pragma unroll
        for (int dt = 0; dt < 8; ++dt) {
            BFrag bv;
#pragma unroll
            for (int j = 0; j < 16; ++j)
                bv.h[j] = (_Float16)Vc[(j < 8 ? g * 8 + j : 16 + g * 8 + (j - 8)) * HD_ + dt * 16 + n];
            o[dt].v = __builtin_amdgcn_wmma_f32_16x16x32_f16(
                false, pa.v, false, bv.v, (short)0, o[dt].v, false, false);
        }
    }

    // Epilogue: normalize and write attn [B,S,NH,HD] (row-major [64,4096])
    float linv[8];
#pragma unroll
    for (int r = 0; r < 8; ++r) linv[r] = 1.0f / lrun[r];
#pragma unroll
    for (int dt = 0; dt < 8; ++dt)
#pragma unroll
        for (int r = 0; r < 8; ++r) {
            const int tok = r + 8 * g;
            attn[(size_t)(b * S_ + tok) * (NH_ * HD_) + qh * HD_ + dt * 16 + n] =
                o[dt].f[r] * linv[r];
        }
}

// ---------------------------------------------------------------------------
// Kernel 4: output projection. attn [64,4096] @ Wo [4096,4096] -> out.
// ---------------------------------------------------------------------------
__global__ __launch_bounds__(128) void oproj_kernel(
    const float* __restrict__ attn, const float* __restrict__ Wo,
    float* __restrict__ out)
{
    const int wid = blockIdx.x * 4 + (threadIdx.x >> 5);   // 0..255
    gemm_strip_64(attn, NH_ * HD_, Wo, H_, wid * 16, out, H_);
}

// ---------------------------------------------------------------------------
extern "C" void kernel_launch(void* const* d_in, const int* in_sizes, int n_in,
                              void* d_out, int out_size, void* d_ws, size_t ws_size,
                              hipStream_t stream)
{
    const float* X   = (const float*)d_in[0];   // hidden_states [4,16,4096]
    const float* kc  = (const float*)d_in[1];   // k_cache [4,8192,8,128]
    const float* vc  = (const float*)d_in[2];   // v_cache
    const float* Wq  = (const float*)d_in[3];   // [4096, 4096]
    const float* Wk  = (const float*)d_in[4];   // [4096, 1024]
    const float* Wv  = (const float*)d_in[5];   // [4096, 1024]
    const float* Wo  = (const float*)d_in[6];   // [4096, 4096]
    (void)in_sizes; (void)n_in; (void)ws_size;

    float* ws    = (float*)d_ws;
    float* Qb    = ws;                          // 64*4096 = 262144
    float* Kb    = ws + 262144;                 // 64*1024 =  65536
    float* Vb    = ws + 327680;                 // 64*1024 =  65536
    float* attnb = ws + 393216;                 // 64*4096 = 262144

    qkv_proj_kernel<<<96, 128, 0, stream>>>(X, Wq, Wk, Wv, Qb, Kb, Vb);
    rope_kernel<<<640, 256, 0, stream>>>(Qb, Kb);
    attn_kernel<<<B_ * NKV_, 128, 0, stream>>>(Qb, Kb, Vb, kc, vc, attnb);
    oproj_kernel<<<64, 128, 0, stream>>>(attnb, Wo, (float*)d_out);
}
